// EnhancedSpatialTransformerEncoder_38062000177743
// MI455X (gfx1250) — matrix operand order repaired and checked by reading, not compiled, not run on hardware
//
#include <hip/hip_runtime.h>
#include <hip/hip_bf16.h>

// ---------------- problem constants ----------------
#define LAYERS 4
#define BATCH  32
#define NN     512          // nodes / seq len
#define DM     256          // d_model
#define HH     8            // heads
#define HD     32           // head dim
#define FF     1024         // ffn dim
#define ROWS   (BATCH*NN)   // 16384 token rows

typedef __attribute__((ext_vector_type(16))) _Float16 v16h;
typedef __attribute__((ext_vector_type(8)))  float    v8f;

__device__ __forceinline__ v8f wmma_f16(v16h a, v16h b, v8f c) {
  // D = A(16x32 f16) * B(32x16 f16) + C(f32)
  return __builtin_amdgcn_wmma_f32_16x16x32_f16(false, a, false, b, (short)0, c,
                                                false, false);
}

__device__ __forceinline__ v8f zero8() {
  v8f z;
#pragma unroll
  for (int i = 0; i < 8; ++i) z[i] = 0.0f;
  return z;
}

// ---- fragment loaders (contiguous k order, k = 16*(lane/16) + element, not the ISA table order; wave32) ----
// A 16x32 f16: lanes 0-15 -> M=lane, K={2v,2v+1};
// lanes 16-31 add K+=16.  src is row-major [16 x >=32] with row stride ld.
__device__ __forceinline__ v16h load_a16x32(const _Float16* __restrict__ s, int ld) {
  int lane = threadIdx.x & 31;
  int m    = lane & 15;
  int koff = (lane >> 4) ? 16 : 0;
  v16h a;
#pragma unroll
  for (int v = 0; v < 8; ++v) {
    int k = 2 * v + koff;
    a[2 * v]     = s[m * ld + k];
    a[2 * v + 1] = s[m * ld + k + 1];
  }
  return a;
}

// B 32x16 f16: VGPR v holds K={2v,2v+1} for lanes 0-15 (N=lane),
// K+=16 for lanes 16-31.  src is row-major [K x N] with row stride ld.
__device__ __forceinline__ v16h load_b32x16(const _Float16* __restrict__ s, int ld) {
  int lane = threadIdx.x & 31;
  int n    = lane & 15;
  int koff = (lane >> 4) ? 16 : 0;
  v16h b;
#pragma unroll
  for (int v = 0; v < 8; ++v) {
    int k = 2 * v + koff;
    b[2 * v]     = s[k * ld + n];
    b[2 * v + 1] = s[(k + 1) * ld + n];
  }
  return b;
}

// B fragment where the source is stored transposed: s is [N x K] row-major
// (e.g. K-matrix stored [key][d]; B(k,n) = s[n*ld + k]).
__device__ __forceinline__ v16h load_bT(const _Float16* __restrict__ s, int ld) {
  int lane = threadIdx.x & 31;
  int n    = lane & 15;
  int koff = (lane >> 4) ? 16 : 0;
  v16h b;
#pragma unroll
  for (int v = 0; v < 8; ++v) {
    int k = 2 * v + koff;
    b[2 * v]     = s[n * ld + k];
    b[2 * v + 1] = s[n * ld + k + 1];
  }
  return b;
}

// =====================================================================
// Tiled WMMA GEMM:  C[M,N] = act( A[M,K] * W[K,N] + bias ) (+res)
// Block tile 128x128, BK=32, 256 threads (8 waves), wave tile 32x64.
// Double-buffered LDS: global loads for tile k+1 issue before the
// WMMAs for tile k, one barrier per k-step (LOADcnt/DScnt decoupling).
// =====================================================================
template <typename TA, bool HAS_BIAS, bool RELU, bool HAS_RES, bool OUT_HALF, bool TRANSB>
__global__ __launch_bounds__(256) void gemm_kernel(
    const TA* __restrict__ A, const float* __restrict__ W,
    const float* __restrict__ bias, const float* __restrict__ res,
    void* __restrict__ Cout, int M, int N, int K) {
  __shared__ _Float16 sA[2][128 * 32];   // 16 KB
  __shared__ _Float16 sB[2][32 * 128];   // 16 KB

  const int tid  = threadIdx.x;
  const int wave = tid >> 5;
  const int lane = tid & 31;
  const int nl   = lane & 15;
  const int hi   = lane >> 4;
  const int wm   = wave & 3;   // 4 waves along M (32 rows each)
  const int wn   = wave >> 2;  // 2 waves along N (64 cols each)
  const int m0   = blockIdx.x * 128;
  const int n0   = blockIdx.y * 128;

  v8f acc[2][4];
#pragma unroll
  for (int i = 0; i < 2; ++i)
#pragma unroll
    for (int j = 0; j < 4; ++j) acc[i][j] = zero8();

  // ---- tile stager ----
  auto stage = [&](int buf, int k0) {
    _Float16* dA = sA[buf];
    _Float16* dB = sB[buf];
    if constexpr (sizeof(TA) == 4) {
#pragma unroll
      for (int t = 0; t < 4; ++t) {
        int lin = tid + t * 256;           // 0..1023 float4 slots (128x32)
        int r   = lin >> 3;
        int c4  = (lin & 7) * 4;
        const float4 f =
            *(const float4*)((const float*)A + (size_t)(m0 + r) * K + k0 + c4);
        dA[r * 32 + c4 + 0] = (_Float16)f.x;
        dA[r * 32 + c4 + 1] = (_Float16)f.y;
        dA[r * 32 + c4 + 2] = (_Float16)f.z;
        dA[r * 32 + c4 + 3] = (_Float16)f.w;
      }
    } else {
#pragma unroll
      for (int t = 0; t < 16; ++t) {
        int lin = tid + t * 256;           // 0..4095 halves
        int r   = lin >> 5;
        int c   = lin & 31;
        dA[r * 32 + c] = ((const _Float16*)A)[(size_t)(m0 + r) * K + k0 + c];
      }
    }
    if constexpr (!TRANSB) {
#pragma unroll
      for (int t = 0; t < 4; ++t) {
        int lin = tid + t * 256;           // 0..1023 float4 slots (32x128)
        int r   = lin >> 5;
        int c4  = (lin & 31) * 4;
        const float4 f = *(const float4*)(W + (size_t)(k0 + r) * N + n0 + c4);
        dB[r * 128 + c4 + 0] = (_Float16)f.x;
        dB[r * 128 + c4 + 1] = (_Float16)f.y;
        dB[r * 128 + c4 + 2] = (_Float16)f.z;
        dB[r * 128 + c4 + 3] = (_Float16)f.w;
      }
    } else {
      // W given as [N x K]; B(k,n) = W[n*K + k]
#pragma unroll
      for (int t = 0; t < 16; ++t) {
        int lin = tid + t * 256;           // 0..4095
        int n   = lin >> 5;
        int k   = lin & 31;
        dB[k * 128 + n] = (_Float16)W[(size_t)(n0 + n) * K + k0 + k];
      }
    }
  };

  const int nk = K >> 5;
  stage(0, 0);
  for (int kk = 0; kk < nk; ++kk) {
    __syncthreads();                       // stage(kk) complete
    const int cur = kk & 1;
    if (kk + 1 < nk) stage(cur ^ 1, (kk + 1) << 5);  // prefetch next tile

    const _Float16* cA = sA[cur];
    const _Float16* cB = sB[cur];
    v16h a0 = load_a16x32(cA + (wm * 32 + 0)  * 32, 32);
    v16h a1 = load_a16x32(cA + (wm * 32 + 16) * 32, 32);
#pragma unroll
    for (int j = 0; j < 4; ++j) {
      v16h b = load_b32x16(cB + wn * 64 + j * 16, 128);
      acc[0][j] = wmma_f16(a0, b, acc[0][j]);
      acc[1][j] = wmma_f16(a1, b, acc[1][j]);
    }
  }

  // ---- epilogue ----
#pragma unroll
  for (int i = 0; i < 2; ++i)
#pragma unroll
    for (int j = 0; j < 4; ++j)
#pragma unroll
      for (int v = 0; v < 8; ++v) {
        int m = m0 + wm * 32 + i * 16 + v + 8 * hi;
        int n = n0 + wn * 64 + j * 16 + nl;
        float c = acc[i][j][v];
        if constexpr (HAS_BIAS) c += bias[n];
        if constexpr (RELU) c = fmaxf(c, 0.0f);
        if constexpr (HAS_RES) c += res[(size_t)m * N + n];
        if constexpr (OUT_HALF)
          ((_Float16*)Cout)[(size_t)m * N + n] = (_Float16)c;
        else
          ((float*)Cout)[(size_t)m * N + n] = c;
      }
}

// =====================================================================
// Flash attention: grid.x = B*H, grid.y = N/128.  8 waves x 16 q-rows.
// S = Q K^T * scale + bias ; online softmax ; O = P V ; O stored f32.
// =====================================================================
__global__ __launch_bounds__(256) void attn_kernel(
    const _Float16* __restrict__ Qg, const _Float16* __restrict__ Kg,
    const _Float16* __restrict__ Vg, const float* __restrict__ bias,
    float* __restrict__ Og) {
  __shared__ _Float16 sQ[128 * 32];
  __shared__ _Float16 sK[128 * 32];
  __shared__ _Float16 sV[128 * 32];
  __shared__ _Float16 sP[8 * 16 * 128];  // per-wave P staging (32 KB)

  const int tid  = threadIdx.x;
  const int wave = tid >> 5;
  const int lane = tid & 31;
  const int nl   = lane & 15;
  const int hi   = lane >> 4;

  const int bh = blockIdx.x;
  const int bb = bh / HH;
  const int h  = bh % HH;
  const int q0 = blockIdx.y * 128;

  const size_t base = ((size_t)bb * NN) * DM + (size_t)h * HD;

  // stage Q tile (128x32)
#pragma unroll
  for (int t = 0; t < 16; ++t) {
    int lin = tid + t * 256;
    int r   = lin >> 5;
    int c   = lin & 31;
    sQ[r * 32 + c] = Qg[base + (size_t)(q0 + r) * DM + c];
  }
  __syncthreads();

  const v16h qa = load_a16x32(sQ + (wave * 16) * 32, 32);

  v8f oacc[2];
  oacc[0] = zero8();
  oacc[1] = zero8();
  float rowM[8], rowL[8];
#pragma unroll
  for (int v = 0; v < 8; ++v) { rowM[v] = -1e30f; rowL[v] = 0.0f; }

  const float scale = 0.17677669529663687f;  // 1/sqrt(32)
  _Float16* __restrict__ pP = sP + wave * 16 * 128;

  for (int kb = 0; kb < NN / 128; ++kb) {
    const int kr0 = kb * 128;
    // stage K/V blocks (128x32 each)
#pragma unroll
    for (int t = 0; t < 16; ++t) {
      int lin = tid + t * 256;
      int r   = lin >> 5;
      int c   = lin & 31;
      sK[r * 32 + c] = Kg[base + (size_t)(kr0 + r) * DM + c];
      sV[r * 32 + c] = Vg[base + (size_t)(kr0 + r) * DM + c];
    }
    __syncthreads();

    // ---- S tiles: 8 WMMAs (K dim = HD = 32, single step each) ----
    v8f sfr[8];
#pragma unroll
    for (int j = 0; j < 8; ++j) {
      v16h bk = load_bT(sK + (j * 16) * 32, 32);  // B(k=d, n=key)
      sfr[j] = wmma_f16(qa, bk, zero8());
    }

    // ---- scale + shared spatial bias + online softmax ----
#pragma unroll
    for (int v = 0; v < 8; ++v) {
      const int qrow = q0 + wave * 16 + v + 8 * hi;
      float mmax = -1e30f;
#pragma unroll
      for (int j = 0; j < 8; ++j) {
        int key = kr0 + j * 16 + nl;
        float s = sfr[j][v] * scale + bias[(size_t)qrow * NN + key];
        sfr[j][v] = s;
        mmax = fmaxf(mmax, s);
      }
      // reduce max over 16 lanes of this half
      mmax = fmaxf(mmax, __shfl_xor(mmax, 1, 32));
      mmax = fmaxf(mmax, __shfl_xor(mmax, 2, 32));
      mmax = fmaxf(mmax, __shfl_xor(mmax, 4, 32));
      mmax = fmaxf(mmax, __shfl_xor(mmax, 8, 32));

      float newM  = fmaxf(rowM[v], mmax);
      float alpha = __expf(rowM[v] - newM);
      rowM[v] = newM;

      float bs = 0.0f;
#pragma unroll
      for (int j = 0; j < 8; ++j) {
        float p = __expf(sfr[j][v] - newM);
        sfr[j][v] = p;
        bs += p;
      }
      bs += __shfl_xor(bs, 1, 32);
      bs += __shfl_xor(bs, 2, 32);
      bs += __shfl_xor(bs, 4, 32);
      bs += __shfl_xor(bs, 8, 32);
      rowL[v] = rowL[v] * alpha + bs;
      oacc[0][v] *= alpha;
      oacc[1][v] *= alpha;
    }

    // ---- P (C-layout f32) -> wave-private LDS as f16 A-source ----
#pragma unroll
    for (int j = 0; j < 8; ++j)
#pragma unroll
      for (int v = 0; v < 8; ++v)
        pP[(v + 8 * hi) * 128 + j * 16 + nl] = (_Float16)sfr[j][v];

    // ---- O += P(16x128) * V(128x32): 4 k-chunks x 2 n-frags ----
#pragma unroll
    for (int kc = 0; kc < 4; ++kc) {
      v16h pa = load_a16x32(pP + kc * 32, 128);
      v16h vb0 = load_b32x16(sV + (kc * 32) * 32 + 0, 32);
      v16h vb1 = load_b32x16(sV + (kc * 32) * 32 + 16, 32);
      oacc[0] = wmma_f16(pa, vb0, oacc[0]);
      oacc[1] = wmma_f16(pa, vb1, oacc[1]);
    }
    __syncthreads();  // before next block overwrites sK/sV
  }

  // ---- normalize and store O (f32, [B,N,D] at head offset) ----
#pragma unroll
  for (int j2 = 0; j2 < 2; ++j2)
#pragma unroll
    for (int v = 0; v < 8; ++v) {
      int qrow = q0 + wave * 16 + v + 8 * hi;
      float o  = oacc[j2][v] / rowL[v];
      Og[((size_t)bb * NN + qrow) * DM + h * HD + j2 * 16 + nl] = o;
    }
}

// =====================================================================
// LayerNorm over last dim (256): one wave per row.
// =====================================================================
__global__ __launch_bounds__(256) void ln_kernel(
    const float* __restrict__ in, const float* __restrict__ g,
    const float* __restrict__ b, float* __restrict__ out) {
  const int wave = threadIdx.x >> 5;
  const int lane = threadIdx.x & 31;
  const int row  = blockIdx.x * 8 + wave;
  const float* x = in + (size_t)row * DM;

  float vals[8];
  float s = 0.0f;
#pragma unroll
  for (int t = 0; t < 8; ++t) {
    vals[t] = x[lane + t * 32];
    s += vals[t];
  }
#pragma unroll
  for (int o = 16; o > 0; o >>= 1) s += __shfl_xor(s, o, 32);
  const float mu = s * (1.0f / DM);

  float vs = 0.0f;
#pragma unroll
  for (int t = 0; t < 8; ++t) {
    float d = vals[t] - mu;
    vs += d * d;
  }
#pragma unroll
  for (int o = 16; o > 0; o >>= 1) vs += __shfl_xor(vs, o, 32);
  const float rinv = rsqrtf(vs * (1.0f / DM) + 1e-5f);

#pragma unroll
  for (int t = 0; t < 8; ++t) {
    int c = lane + t * 32;
    out[(size_t)row * DM + c] = (vals[t] - mu) * rinv * g[c] + b[c];
  }
}

// =====================================================================
// Host orchestration
// =====================================================================
extern "C" void kernel_launch(void* const* d_in, const int* in_sizes, int n_in,
                              void* d_out, int out_size, void* d_ws, size_t ws_size,
                              hipStream_t stream) {
  const float* src = (const float*)d_in[0];
  const float* e1  = (const float*)d_in[1];
  const float* e2  = (const float*)d_in[2];
  const float* Wq  = (const float*)d_in[3];
  const float* bq  = (const float*)d_in[4];
  const float* Wk  = (const float*)d_in[5];
  const float* bk  = (const float*)d_in[6];
  const float* Wv  = (const float*)d_in[7];
  const float* bv  = (const float*)d_in[8];
  const float* Wo  = (const float*)d_in[9];
  const float* bo  = (const float*)d_in[10];
  const float* W1  = (const float*)d_in[11];
  const float* b1  = (const float*)d_in[12];
  const float* W2  = (const float*)d_in[13];
  const float* b2  = (const float*)d_in[14];
  const float* g1  = (const float*)d_in[15];
  const float* be1 = (const float*)d_in[16];
  const float* g2  = (const float*)d_in[17];
  const float* be2 = (const float*)d_in[18];

  // ---- workspace carve-up ----
  char* w = (char*)d_ws;
  float*     x    = (float*)w;      w += (size_t)ROWS * DM * 4;  // 16 MB activations
  float*     tmp  = (float*)w;      w += (size_t)ROWS * DM * 4;  // 16 MB pre-LN
  float*     ao   = (float*)w;      w += (size_t)ROWS * DM * 4;  // 16 MB attn out
  _Float16*  qh   = (_Float16*)w;   w += (size_t)ROWS * DM * 2;  //  8 MB
  _Float16*  kh   = (_Float16*)w;   w += (size_t)ROWS * DM * 2;  //  8 MB
  _Float16*  vh   = (_Float16*)w;   w += (size_t)ROWS * DM * 2;  //  8 MB
  _Float16*  hid  = (_Float16*)w;   w += (size_t)ROWS * FF * 2;  // 32 MB ffn hidden
  float*     bsp  = (float*)w;      w += (size_t)NN * NN * 4;    //  1 MB spatial bias

  // shared spatial bias = e1 @ e2^T   (M=512, N=512, K=256)
  gemm_kernel<float, false, false, false, false, true>
      <<<dim3(NN / 128, NN / 128), 256, 0, stream>>>(e1, e2, nullptr, nullptr, bsp,
                                                     NN, NN, DM);

  // x = src
  hipMemcpyAsync(x, src, (size_t)ROWS * DM * 4, hipMemcpyDeviceToDevice, stream);

  const dim3 gProj(ROWS / 128, DM / 128);   // 128 x 2
  const dim3 gFfn1(ROWS / 128, FF / 128);   // 128 x 8
  const dim3 gAttn(BATCH * HH, NN / 128);   // 256 x 4

  for (int l = 0; l < LAYERS; ++l) {
    const float* Wql = Wq + (size_t)l * DM * DM;
    const float* Wkl = Wk + (size_t)l * DM * DM;
    const float* Wvl = Wv + (size_t)l * DM * DM;
    const float* Wol = Wo + (size_t)l * DM * DM;
    const float* W1l = W1 + (size_t)l * DM * FF;
    const float* W2l = W2 + (size_t)l * FF * DM;

    // Q/K/V projections -> f16
    gemm_kernel<float, true, false, false, true, false>
        <<<gProj, 256, 0, stream>>>(x, Wql, bq + l * DM, nullptr, qh, ROWS, DM, DM);
    gemm_kernel<float, true, false, false, true, false>
        <<<gProj, 256, 0, stream>>>(x, Wkl, bk + l * DM, nullptr, kh, ROWS, DM, DM);
    gemm_kernel<float, true, false, false, true, false>
        <<<gProj, 256, 0, stream>>>(x, Wvl, bv + l * DM, nullptr, vh, ROWS, DM, DM);

    // flash attention with shared spatial bias
    attn_kernel<<<gAttn, 256, 0, stream>>>(qh, kh, vh, bsp, ao);

    // O projection + residual, then LN1 -> x
    gemm_kernel<float, true, false, true, false, false>
        <<<gProj, 256, 0, stream>>>(ao, Wol, bo + l * DM, x, tmp, ROWS, DM, DM);
    ln_kernel<<<ROWS / 8, 256, 0, stream>>>(tmp, g1 + l * DM, be1 + l * DM, x);

    // FFN1 (ReLU, f16 hidden)
    gemm_kernel<float, true, true, false, true, false>
        <<<gFfn1, 256, 0, stream>>>(x, W1l, b1 + l * FF, nullptr, hid, ROWS, FF, DM);
    // FFN2 (+ residual), then LN2 -> x (last layer writes d_out)
    gemm_kernel<_Float16, true, false, true, false, false>
        <<<gProj, 256, 0, stream>>>(hid, W2l, b2 + l * DM, x, tmp, ROWS, DM, FF);
    float* lnOut = (l == LAYERS - 1) ? (float*)d_out : x;
    ln_kernel<<<ROWS / 8, 256, 0, stream>>>(tmp, g2 + l * DM, be2 + l * DM, lnOut);
  }
}
